// PlanetWarsAgentGNN_30305289241205
// MI455X (gfx1250) — matrix operand order repaired and checked by reading, not compiled, not run on hardware
//
#include <hip/hip_runtime.h>

typedef __attribute__((ext_vector_type(16))) _Float16 v16h;
typedef __attribute__((ext_vector_type(8)))  float    v8f;

#define GB   512   // graphs
#define PP   32    // nodes per graph
#define FF   16    // input node features
#define HH   4     // heads
#define DD   128   // hidden = H*C
#define EPG  992   // edges per graph = P*(P-1)
#define OUTC 34    // P+2 outputs per graph

// Workspace layout (in _Float16 units): pre-swizzled B-fragment weights.
// Per matrix: [chunk][nt(8)][lane(32)][slot(16)] halves.
#define WS_W1L 0            // 1 chunk  -> 4096 halves
#define WS_W1R 4096         // 1 chunk  -> 4096
#define WS_W2L 8192         // 4 chunks -> 16384
#define WS_W2R 24576        // 4 chunks -> 16384
#define WS_WS1 40960        // 4 chunks -> 16384
#define WS_TOT 57344        // 112 KB

// ---------------------------------------------------------------------------
// Fragment maps (ISA 7.12.2, wave32, 16-bit operands, f32 16x16 C/D):
//   A (16x32): lane = m + 16*((k%16)>=8), slot = (k%8) + 8*(k>=16)
//     inverse: m = lane%16, k = (slot&7) + 8*(lane>>4) + 16*(slot>>3)
//   B (32x16): lane = n + 16*((k%16)>=8), slot = (k%8) + 8*(k>=16)
//     inverse: n = lane%16, k = (slot&7) + 8*(lane>>4) + 16*(slot>>3)
//   D (16x16): row = v + 8*(lane>>4), col = lane%16
// ---------------------------------------------------------------------------

// Convert f32 row-major W[128][Kreal] -> f16 B-fragment layout (zero-padded).
__global__ __launch_bounds__(256)
void prep_weights_kernel(const float* __restrict__ W1l, const float* __restrict__ W1r,
                         const float* __restrict__ W2l, const float* __restrict__ W2r,
                         const float* __restrict__ Ws1, _Float16* __restrict__ wsh)
{
  const int idx = (int)(blockIdx.x * 256 + threadIdx.x);
  if (idx >= WS_TOT) return;
  const float* W; int Kreal; int base;
  if      (idx < WS_W1R) { W = W1l; Kreal = FF;  base = WS_W1L; }
  else if (idx < WS_W2L) { W = W1r; Kreal = FF;  base = WS_W1R; }
  else if (idx < WS_W2R) { W = W2l; Kreal = DD;  base = WS_W2L; }
  else if (idx < WS_WS1) { W = W2r; Kreal = DD;  base = WS_W2R; }
  else                   { W = Ws1; Kreal = DD;  base = WS_WS1; }
  const int r     = idx - base;
  const int slot  = r & 15;
  const int lane  = (r >> 4) & 31;
  const int nt    = (r >> 9) & 7;
  const int chunk = r >> 12;
  const int n = nt * 16 + (lane & 15);
  const int k = chunk * 32 + (slot & 7) + 8 * (lane >> 4) + 16 * (slot >> 3);
  wsh[idx] = (_Float16)((k < Kreal) ? W[n * Kreal + k] : 0.f);
}

// Fused l/r projection: outL/outR[32][128] = A @ B{l,r} + b{l,r}.
// A fragments shared; two independent WMMA chains interleave to hide the
// D->C accumulation hazard. All fragment loads are hoisted so the compiler
// can clause them and issue a single wait before the WMMA chain.
template<int NCHUNK>
__device__ __forceinline__ void dual_linear_wmma_32xD(
    const v16h* __restrict__ wsBl, const v16h* __restrict__ wsBr,
    const float* __restrict__ bl,  const float* __restrict__ br,
    const v16h* sAF,                     // [NCHUNK][2][32] v16h, LDS
    float* outL, float* outR,            // LDS [32][128] f32
    int wave, int lane)
{
  const int m  = lane & 15;
  const int hi = lane >> 4;
  #pragma unroll
  for (int q = 0; q < 2; ++q) {
    const int tile = wave * 2 + q;
    const int mt = tile >> 3;
    const int nt = tile & 7;
    v16h a[NCHUNK], bL[NCHUNK], bR[NCHUNK];
    #pragma unroll
    for (int c = 0; c < NCHUNK; ++c) {
      a[c]  = sAF [(c * 2 + mt) * 32 + lane];   // ds_load_b128 x2
      bL[c] = wsBl[(c * 8 + nt) * 32 + lane];   // global_load_b128 x2
      bR[c] = wsBr[(c * 8 + nt) * 32 + lane];
    }
    v8f accL = {}, accR = {};
    #pragma unroll
    for (int c = 0; c < NCHUNK; ++c) {
      accL = __builtin_amdgcn_wmma_f32_16x16x32_f16(
          false, a[c], false, bL[c], (short)0, accL, false, false);
      accR = __builtin_amdgcn_wmma_f32_16x16x32_f16(
          false, a[c], false, bR[c], (short)0, accR, false, false);
    }
    const float bvL = bl[nt * 16 + m];
    const float bvR = br[nt * 16 + m];
    const int   ob  = (mt * 16 + 8 * hi) * DD + nt * 16 + m;
    #pragma unroll
    for (int v = 0; v < 8; ++v) {
      outL[ob + v * DD] = accL[v] + bvL;
      outR[ob + v * DD] = accR[v] + bvR;
    }
  }
}

// Single-output variant (source-actor hidden layer, with ReLU).
template<int NCHUNK>
__device__ __forceinline__ void linear_wmma_32xD(
    const v16h* __restrict__ wsB, const float* __restrict__ bias,
    const v16h* sAF, float* out, int wave, int lane, bool relu)
{
  const int m  = lane & 15;
  const int hi = lane >> 4;
  #pragma unroll
  for (int q = 0; q < 2; ++q) {
    const int tile = wave * 2 + q;
    const int mt = tile >> 3;
    const int nt = tile & 7;
    v16h a[NCHUNK], b[NCHUNK];
    #pragma unroll
    for (int c = 0; c < NCHUNK; ++c) {
      a[c] = sAF[(c * 2 + mt) * 32 + lane];
      b[c] = wsB[(c * 8 + nt) * 32 + lane];
    }
    v8f acc = {};
    #pragma unroll
    for (int c = 0; c < NCHUNK; ++c)
      acc = __builtin_amdgcn_wmma_f32_16x16x32_f16(
          false, a[c], false, b[c], (short)0, acc, false, false);
    const float bv = bias[nt * 16 + m];
    #pragma unroll
    for (int v = 0; v < 8; ++v) {
      float val = acc[v] + bv;
      if (relu) val = fmaxf(val, 0.f);
      out[(mt * 16 + 8 * hi + v) * DD + nt * 16 + m] = val;
    }
  }
}

// ---------------------------------------------------------------------------
// One workgroup per graph: dense GATv2 x2 + norm + pool + actor/critic heads.
// ---------------------------------------------------------------------------
__global__ __launch_bounds__(256)
void planetwars_gnn_kernel(
    const float* __restrict__ x, const float* __restrict__ edge_attr,
    const _Float16* __restrict__ wsh,
    const float* __restrict__ b1l, const float* __restrict__ b1r,
    const float* __restrict__ W1e, const float* __restrict__ att1,
    const float* __restrict__ bias1,
    const float* __restrict__ b2l, const float* __restrict__ b2r,
    const float* __restrict__ W2e, const float* __restrict__ att2,
    const float* __restrict__ bias2,
    const float* __restrict__ Wc1, const float* __restrict__ bc1,
    const float* __restrict__ Wc2, const float* __restrict__ bc2,
    const float* __restrict__ bs1,
    const float* __restrict__ Ws2, const float* __restrict__ bs2,
    const float* __restrict__ Wn1, const float* __restrict__ bn1,
    const float* __restrict__ Wn2, const float* __restrict__ bn2,
    float* __restrict__ out)
{
  __shared__ float sXL[PP * DD];        // 16 KB  source-side projection
  __shared__ float sXR[PP * DD];        // 16 KB  dest-side projection / layer output
  __shared__ float sHS[PP * DD];        // 16 KB  UNION: hidden h  <->  score[i][j][h]
  __shared__ v16h  sAF[4 * 2 * 32];     // 8 KB   A-fragments (f16, swizzled)
  __shared__ float sWe[DD * 5];         // 2.5 KB edge-weight matrix (staged)
  __shared__ float sAtt[DD];            // attention vector (staged)
  __shared__ float sVec[DD];            // mean / pooled g
  __shared__ float sT2[2 * DD];         // critic + noop hidden

  float*     sH     = sHS;
  float*     sScore = sHS;
  _Float16*  af     = (_Float16*)sAF;

  const int tid  = (int)threadIdx.x;
  const int wave = tid >> 5;
  const int lane = tid & 31;
  const int g    = (int)blockIdx.x;

  const float* ea_base = edge_attr + (size_t)g * EPG * 5;
  __builtin_prefetch(ea_base + (size_t)tid * 5, 0, 1);   // global_prefetch_b8

  // ======================= two GATv2 layers =======================
  for (int layer = 0; layer < 2; ++layer) {
    const float* bl  = layer ? b2l   : b1l;
    const float* br  = layer ? b2r   : b1r;
    const float* We  = layer ? W2e   : W1e;
    const float* att = layer ? att2  : att1;
    const float* bia = layer ? bias2 : bias1;
    const v16h* wBl  = (const v16h*)(wsh + (layer ? WS_W2L : WS_W1L));
    const v16h* wBr  = (const v16h*)(wsh + (layer ? WS_W2R : WS_W1R));

    // ---- build A fragments (f16, swizzled) ----
    if (layer == 0) {
      // from global x [32][16], zero-padded to K=32 (1 chunk)
      const float* xg = x + (size_t)g * PP * FF;
      for (int idx = tid; idx < 2 * 32 * 16; idx += 256) {
        const int slot = idx & 15, ln = (idx >> 4) & 31, mt = (idx >> 9) & 1;
        const int mrow = mt * 16 + (ln & 15);
        const int k = (slot & 7) + 8 * (ln >> 4) + 16 * (slot >> 3);
        af[idx] = (_Float16)((k < FF) ? xg[mrow * FF + k] : 0.f);
      }
    } else {
      // from sH [32][128] (4 chunks)
      for (int idx = tid; idx < 4 * 2 * 32 * 16; idx += 256) {
        const int slot = idx & 15, ln = (idx >> 4) & 31;
        const int mt = (idx >> 9) & 1, chunk = idx >> 10;
        const int mrow = mt * 16 + (ln & 15);
        const int k = chunk * 32 + (slot & 7) + 8 * (ln >> 4) + 16 * (slot >> 3);
        af[idx] = (_Float16)sH[mrow * DD + k];
      }
    }
    // stage We [128,5] and att [128]
    for (int i = tid; i < DD * 5; i += 256) sWe[i] = We[i];
    if (tid < DD) sAtt[tid] = att[tid];
    __syncthreads();

    // xl = in @ Wl^T + bl ; xr = in @ Wr^T + br      (fused dual WMMA)
    if (layer == 0)
      dual_linear_wmma_32xD<1>(wBl, wBr, bl, br, sAF, sXL, sXR, wave, lane);
    else
      dual_linear_wmma_32xD<4>(wBl, wBr, bl, br, sAF, sXL, sXR, wave, lane);
    __syncthreads();   // sH / sAF consumed -> sScore region reusable

    // per-edge attention scores: score[i][j][h]
    for (int pair = tid; pair < PP * PP; pair += 256) {
      const int i = pair >> 5;   // src
      const int j = pair & 31;   // dst
      if (i == j) {
        #pragma unroll
        for (int h = 0; h < HH; ++h) sScore[pair * HH + h] = -1e30f;
      } else {
        const int el = i * 31 + j - (j > i ? 1 : 0);
        const float* ea = ea_base + el * 5;
        const float e0 = ea[0], e1 = ea[1], e2 = ea[2], e3 = ea[3], e4 = ea[4];
        float acch[HH] = {0.f, 0.f, 0.f, 0.f};
        #pragma unroll 4
        for (int c = 0; c < DD; ++c) {
          const float* w = &sWe[c * 5];
          float ee = fmaf(e0, w[0], fmaf(e1, w[1],
                     fmaf(e2, w[2], fmaf(e3, w[3], e4 * w[4]))));
          float t  = sXL[i * DD + c] + sXR[j * DD + c] + ee;
          float gg = (t > 0.f) ? t : 0.2f * t;          // leaky_relu(0.2)
          acch[c >> 5] = fmaf(gg, sAtt[c], acch[c >> 5]);
        }
        #pragma unroll
        for (int h = 0; h < HH; ++h) sScore[pair * HH + h] = acch[h];
      }
    }
    __syncthreads();

    // softmax over src i for each (dst j, head h)
    if (tid < PP * HH) {
      const int j = tid >> 2, h = tid & 3;
      float mx = -1e30f;
      for (int i = 0; i < PP; ++i)
        mx = fmaxf(mx, sScore[(i * PP + j) * HH + h]);
      float sum = 0.f;
      for (int i = 0; i < PP; ++i) {
        float ex = expf(sScore[(i * PP + j) * HH + h] - mx);
        sScore[(i * PP + j) * HH + h] = ex;
        sum += ex;
      }
      const float inv = 1.f / (sum + 1e-16f);
      for (int i = 0; i < PP; ++i) sScore[(i * PP + j) * HH + h] *= inv;
    }
    __syncthreads();

    // aggregate: out[j][d] = sum_i alpha[i][j][h(d)] * xl[i][d] + bias[d]
    for (int idx = tid; idx < PP * DD; idx += 256) {
      const int j = idx >> 7, d = idx & 127, h = d >> 5;
      float acc = bia[d];
      for (int i = 0; i < PP; ++i)
        acc = fmaf(sScore[(i * PP + j) * HH + h], sXL[i * DD + d], acc);
      sXR[idx] = acc;
    }
    __syncthreads();   // sScore dead -> sH region reusable

    if (layer == 0) {
      // MeanSubtractionNorm per graph, then ReLU -> sH
      if (tid < DD) {
        float s = 0.f;
        for (int p = 0; p < PP; ++p) s += sXR[p * DD + tid];
        sVec[tid] = s * (1.f / PP);
      }
      __syncthreads();
      for (int idx = tid; idx < PP * DD; idx += 256)
        sH[idx] = fmaxf(sXR[idx] - sVec[idx & 127], 0.f);
      __syncthreads();
    } else {
      // ReLU -> sH, then global mean pool -> sVec
      for (int idx = tid; idx < PP * DD; idx += 256)
        sH[idx] = fmaxf(sXR[idx], 0.f);
      __syncthreads();
      if (tid < DD) {
        float s = 0.f;
        for (int p = 0; p < PP; ++p) s += sH[p * DD + tid];
        sVec[tid] = s * (1.f / PP);
      }
      __syncthreads();
    }
  }

  // ======================= heads =======================
  // rebuild A-fragments from final hidden, then sXL = relu(h @ Ws1^T + bs1)
  for (int idx = tid; idx < 4 * 2 * 32 * 16; idx += 256) {
    const int slot = idx & 15, ln = (idx >> 4) & 31;
    const int mt = (idx >> 9) & 1, chunk = idx >> 10;
    const int mrow = mt * 16 + (ln & 15);
    const int k = chunk * 32 + (slot & 7) + 8 * (ln >> 4) + 16 * (slot >> 3);
    af[idx] = (_Float16)sH[mrow * DD + k];
  }
  __syncthreads();
  linear_wmma_32xD<4>((const v16h*)(wsh + WS_WS1), bs1, sAF, sXL, wave, lane, true);

  // critic / noop hidden on pooled g (sVec): 256 threads = 2 x 128 rows
  {
    const int d = tid & 127;
    const float* Wrow = (tid < DD ? Wc1 : Wn1) + d * DD;
    float s = (tid < DD ? bc1 : bn1)[d];
    for (int c = 0; c < DD; ++c) s = fmaf(Wrow[c], sVec[c], s);
    sT2[tid] = fmaxf(s, 0.f);
  }
  __syncthreads();

  float* orow = out + (size_t)g * OUTC;
  if (tid == 0) {                         // value
    float s = bc2[0];
    for (int c = 0; c < DD; ++c) s = fmaf(Wc2[c], sT2[c], s);
    orow[0] = s;
  } else if (tid == 1) {                  // noop
    float s = bn2[0];
    for (int c = 0; c < DD; ++c) s = fmaf(Wn2[c], sT2[DD + c], s);
    orow[1] = s;
  } else if (tid >= 32 && tid < 32 + PP) {  // per-node source logits
    const int p = tid - 32;
    float s = bs2[0];
    for (int c = 0; c < DD; ++c) s = fmaf(Ws2[c], sXL[p * DD + c], s);
    orow[2 + p] = s;
  }
}

// ---------------------------------------------------------------------------
extern "C" void kernel_launch(void* const* d_in, const int* in_sizes, int n_in,
                              void* d_out, int out_size, void* d_ws, size_t ws_size,
                              hipStream_t stream) {
  (void)in_sizes; (void)n_in; (void)ws_size; (void)out_size;
  // setup_inputs() order:
  // 0:x 1:edge_index 2:edge_attr 3:W1l 4:b1l 5:W1r 6:b1r 7:W1e 8:att1 9:bias1
  // 10:W2l 11:b2l 12:W2r 13:b2r 14:W2e 15:att2 16:bias2
  // 17:Wc1 18:bc1 19:Wc2 20:bc2 21:Ws1 22:bs1 23:Ws2 24:bs2 25:Wn1 26:bn1 27:Wn2 28:bn2
  _Float16* wsh = (_Float16*)d_ws;

  prep_weights_kernel<<<(WS_TOT + 255) / 256, 256, 0, stream>>>(
      (const float*)d_in[3],  (const float*)d_in[5],
      (const float*)d_in[10], (const float*)d_in[12],
      (const float*)d_in[21], wsh);

  planetwars_gnn_kernel<<<GB, 256, 0, stream>>>(
      (const float*)d_in[0],  (const float*)d_in[2],  wsh,
      (const float*)d_in[4],  (const float*)d_in[6],
      (const float*)d_in[7],  (const float*)d_in[8],  (const float*)d_in[9],
      (const float*)d_in[11], (const float*)d_in[13],
      (const float*)d_in[14], (const float*)d_in[15], (const float*)d_in[16],
      (const float*)d_in[17], (const float*)d_in[18],
      (const float*)d_in[19], (const float*)d_in[20],
      (const float*)d_in[22],
      (const float*)d_in[23], (const float*)d_in[24],
      (const float*)d_in[25], (const float*)d_in[26],
      (const float*)d_in[27], (const float*)d_in[28],
      (float*)d_out);
}
